// MultiHeadAttentionLayer_74311524156003
// MI455X (gfx1250) — matrix operand order repaired and checked by reading, not compiled, not run on hardware
//
#include <hip/hip_runtime.h>

#define EMBED  1024
#define NHEADS 16
#define HDIM   64
#define BATCH  2
#define SEQ    2048
#define NTOK   (BATCH * SEQ)          // 4096 rows for projections
#define NE     (NTOK * EMBED)         // 4,194,304 activation elems
#define WE     (EMBED * EMBED)        // 1,048,576 weight elems

typedef __attribute__((ext_vector_type(16))) __bf16 v16bf;
typedef __attribute__((ext_vector_type(8)))  __bf16 v8bf;
typedef __attribute__((ext_vector_type(4)))  __bf16 v4bf;
typedef __attribute__((ext_vector_type(8)))  float  v8f;

// ---------------------------------------------------------------- utilities

__device__ __forceinline__ __bf16 f2bf(float f) {
  // round-to-nearest-even fp32 -> bf16
  unsigned u = __builtin_bit_cast(unsigned, f);
  unsigned r = (u + 0x7FFFu + ((u >> 16) & 1u)) >> 16;
  unsigned short s = (unsigned short)r;
  return __builtin_bit_cast(__bf16, s);
}

// A-fragment (16x32 bf16, M x K): lane m = lane&15.
// lanes 0-15 hold K=0..7 (elems 0-7) and K=16..23 (elems 8-15);
// lanes 16-31 hold K=8..15 and K=24..31.  p points at the 32-elem K chunk
// of this lane's row.
__device__ __forceinline__ v16bf load_a_frag(const __bf16* p, bool hi) {
  const int off = hi ? 8 : 0;
  v8bf lo = *(const v8bf*)(p + off);
  v8bf hh = *(const v8bf*)(p + off + 16);
  v16bf a;
#pragma unroll
  for (int i = 0; i < 8; ++i) { a[i] = lo[i]; a[i + 8] = hh[i]; }
  return a;
}

// B-fragment (32x16 bf16, K x N): lane n = lane&15; lanes 0-15 hold K=0..7
// and K=16..23, lanes 16-31 hold K=8..15 and K=24..31.  p points at K-chunk
// start of this lane's column source row (stored K-contiguous).
__device__ __forceinline__ v16bf load_b_frag(const __bf16* p, bool hi) {
  const int off = hi ? 8 : 0;
  v8bf lo = *(const v8bf*)(p + off);
  v8bf hh = *(const v8bf*)(p + off + 16);
  v16bf b;
#pragma unroll
  for (int i = 0; i < 8; ++i) { b[i] = lo[i]; b[i + 8] = hh[i]; }
  return b;
}

__device__ __forceinline__ v8f wmma_bf16(v16bf a, v16bf b, v8f c) {
  return __builtin_amdgcn_wmma_f32_16x16x32_bf16(
      /*neg_a=*/false, a, /*neg_b=*/false, b,
      /*c_mod=*/(short)0, c, /*reuse_a=*/false, /*reuse_b=*/false);
}

// ------------------------------------------------------- fp32 -> bf16 convert

__global__ __launch_bounds__(256)
void cvt_f32_to_bf16_x4(const float* __restrict__ in, __bf16* __restrict__ out,
                        int n4) {
  int i = blockIdx.x * blockDim.x + threadIdx.x;
  if (i >= n4) return;
  const float4 v = ((const float4*)in)[i];
  v4bf o;
  o[0] = f2bf(v.x); o[1] = f2bf(v.y); o[2] = f2bf(v.z); o[3] = f2bf(v.w);
  ((v4bf*)out)[i] = o;
}

// ------------------------------------------------------------- WMMA GEMM
// C[m,n] = sum_k X[m,k] * W[n,k] + bias[n]   (nn.Linear: x @ W.T + b)
// Block: 256 thr = 8 waves -> 64(M) x 256(N) tile; wave -> 32x64 (8 WMMA/chunk).
// mode 0: store bf16 [B,H,T,hd]        (Q, K)
// mode 1: store bf16 [B,H,hd,T]        (V transposed for PV B-frags)
// mode 2: store f32  [M,N]             (final out-projection)
__global__ __launch_bounds__(256)
void gemm_bf16_wmma(const __bf16* __restrict__ X, const __bf16* __restrict__ W,
                    const float* __restrict__ bias, void* __restrict__ out,
                    int M, int N, int K, int mode) {
  const int lane = threadIdx.x & 31;
  const int w    = threadIdx.x >> 5;
  const int wm   = w & 1;          // 2 wave-rows
  const int wn   = w >> 1;         // 4 wave-cols
  const int m0   = blockIdx.y * 64 + wm * 32;
  const int n0   = blockIdx.x * 256 + wn * 64;
  const int lrow = lane & 15;
  const bool hi  = lane >= 16;

  v8f acc[2][4] = {};

  const __bf16* xr0 = X + (size_t)(m0 + lrow) * K;
  const __bf16* xr1 = X + (size_t)(m0 + 16 + lrow) * K;
  const __bf16* wr[4];
#pragma unroll
  for (int ns = 0; ns < 4; ++ns)
    wr[ns] = W + (size_t)(n0 + ns * 16 + lrow) * K;

  for (int kb = 0; kb < K; kb += 32) {
    v16bf a0 = load_a_frag(xr0 + kb, hi);
    v16bf a1 = load_a_frag(xr1 + kb, hi);
#pragma unroll
    for (int ns = 0; ns < 4; ++ns) {
      v16bf b = load_b_frag(wr[ns] + kb, hi);
      acc[0][ns] = wmma_bf16(a0, b, acc[0][ns]);
      acc[1][ns] = wmma_bf16(a1, b, acc[1][ns]);
    }
  }

#pragma unroll
  for (int ms = 0; ms < 2; ++ms) {
#pragma unroll
    for (int ns = 0; ns < 4; ++ns) {
#pragma unroll
      for (int r = 0; r < 8; ++r) {
        const int m = m0 + ms * 16 + r + (hi ? 8 : 0);
        const int n = n0 + ns * 16 + lrow;
        const float val = acc[ms][ns][r] + bias[n];
        if (mode == 0) {
          const int bb = m / SEQ, tt = m % SEQ;
          const int hh = n / HDIM, dd = n % HDIM;
          ((__bf16*)out)[(((size_t)(bb * NHEADS + hh) * SEQ) + tt) * HDIM + dd] =
              f2bf(val);
        } else if (mode == 1) {
          const int bb = m / SEQ, tt = m % SEQ;
          const int hh = n / HDIM, dd = n % HDIM;
          ((__bf16*)out)[(((size_t)(bb * NHEADS + hh) * HDIM) + dd) * SEQ + tt] =
              f2bf(val);
        } else {
          ((float*)out)[(size_t)m * N + n] = val;
        }
      }
    }
  }
}

// --------------------------------------------------------- flash attention
// Q,K: [B,H,T,hd] bf16   V: [B,H,hd,T] bf16   O: [B*T, E] bf16 (merged heads)
// Block: 128 thr = 4 waves; each wave owns TWO 16-row Q tiles (32 q rows) so
// every K/V fragment feeds 2 WMMAs; keys stream in chunks of 32.
__global__ __launch_bounds__(128)
void flash_attn_bf16(const __bf16* __restrict__ Q, const __bf16* __restrict__ Km,
                     const __bf16* __restrict__ Vt, __bf16* __restrict__ Out,
                     float scale) {
  __shared__ __bf16 plds[4][2][16 * 32];   // per-wave, per-mtile P staging

  const int lane = threadIdx.x & 31;
  const int w    = threadIdx.x >> 5;
  const int bh   = blockIdx.y;
  const int b    = bh / NHEADS;
  const int h    = bh % NHEADS;
  const int q0   = blockIdx.x * 128 + w * 32;
  const int lrow = lane & 15;
  const bool hi  = lane >= 16;

  const __bf16* qbase = Q  + (size_t)bh * SEQ * HDIM;
  const __bf16* kbase = Km + (size_t)bh * SEQ * HDIM;
  const __bf16* vbase = Vt + (size_t)bh * HDIM * SEQ;

  // Q fragments, register-resident for whole key loop: [mtile][k-chunk]
  v16bf qf[2][2];
#pragma unroll
  for (int mt = 0; mt < 2; ++mt) {
    const __bf16* qr = qbase + (size_t)(q0 + mt * 16 + lrow) * HDIM;
    qf[mt][0] = load_a_frag(qr + 0,  hi);
    qf[mt][1] = load_a_frag(qr + 32, hi);
  }

  v8f o[2][4] = {};
  v8f mrow[2], lsum[2];
#pragma unroll
  for (int mt = 0; mt < 2; ++mt)
#pragma unroll
    for (int r = 0; r < 8; ++r) { mrow[mt][r] = -__builtin_inff(); lsum[mt][r] = 0.0f; }

  for (int jb = 0; jb < SEQ; jb += 32) {
    // ---- K fragments for this 32-key chunk (shared by both q tiles)
    v16bf kb00 = load_b_frag(kbase + (size_t)(jb + lrow) * HDIM + 0,       hi);
    v16bf kb01 = load_b_frag(kbase + (size_t)(jb + lrow) * HDIM + 32,      hi);
    v16bf kb10 = load_b_frag(kbase + (size_t)(jb + 16 + lrow) * HDIM + 0,  hi);
    v16bf kb11 = load_b_frag(kbase + (size_t)(jb + 16 + lrow) * HDIM + 32, hi);

#pragma unroll
    for (int mt = 0; mt < 2; ++mt) {
      // S = Q K^T  (two 16x16 C-tiles covering the 32 keys)
      v8f s0 = {}, s1 = {};
      s0 = wmma_bf16(qf[mt][0], kb00, s0);
      s0 = wmma_bf16(qf[mt][1], kb01, s0);
      s1 = wmma_bf16(qf[mt][0], kb10, s1);
      s1 = wmma_bf16(qf[mt][1], kb11, s1);
#pragma unroll
      for (int r = 0; r < 8; ++r) { s0[r] *= scale; s1[r] *= scale; }

      // online softmax: row stats live per-VGPR, columns per-lane (16-wide)
      v8f tmax;
#pragma unroll
      for (int r = 0; r < 8; ++r) tmax[r] = fmaxf(s0[r], s1[r]);
#pragma unroll
      for (int off = 8; off > 0; off >>= 1)
#pragma unroll
        for (int r = 0; r < 8; ++r)
          tmax[r] = fmaxf(tmax[r], __shfl_xor(tmax[r], off, 32));

      v8f cor;
#pragma unroll
      for (int r = 0; r < 8; ++r) {
        const float mnew = fmaxf(mrow[mt][r], tmax[r]);
        cor[r]      = __expf(mrow[mt][r] - mnew);
        s0[r]       = __expf(s0[r] - mnew);
        s1[r]       = __expf(s1[r] - mnew);
        mrow[mt][r] = mnew;
      }
      v8f rsum;
#pragma unroll
      for (int r = 0; r < 8; ++r) rsum[r] = s0[r] + s1[r];
#pragma unroll
      for (int off = 8; off > 0; off >>= 1)
#pragma unroll
        for (int r = 0; r < 8; ++r)
          rsum[r] += __shfl_xor(rsum[r], off, 32);
#pragma unroll
      for (int r = 0; r < 8; ++r) lsum[mt][r] = lsum[mt][r] * cor[r] + rsum[r];
#pragma unroll
      for (int g = 0; g < 4; ++g)
#pragma unroll
        for (int r = 0; r < 8; ++r) o[mt][g][r] *= cor[r];

      // C-layout P (16x32 f32) -> A-layout bf16 via per-wave LDS
      __bf16* pw = plds[w][mt];
#pragma unroll
      for (int r = 0; r < 8; ++r) {
        const int mm = r + (hi ? 8 : 0);
        pw[mm * 32 + lrow]      = f2bf(s0[r]);
        pw[mm * 32 + 16 + lrow] = f2bf(s1[r]);
      }
    }

    __builtin_amdgcn_wave_barrier();
    asm volatile("s_wait_dscnt 0" ::: "memory");
    v16bf pf[2];
#pragma unroll
    for (int mt = 0; mt < 2; ++mt)
      pf[mt] = load_a_frag(plds[w][mt] + lrow * 32, hi);
    __builtin_amdgcn_wave_barrier();

    // ---- O += P V : each V fragment feeds both q tiles
#pragma unroll
    for (int g = 0; g < 4; ++g) {
      v16bf vb = load_b_frag(vbase + (size_t)(g * 16 + lrow) * SEQ + jb, hi);
      o[0][g] = wmma_bf16(pf[0], vb, o[0][g]);
      o[1][g] = wmma_bf16(pf[1], vb, o[1][g]);
    }
  }

  // ---- epilogue: O / l, merge heads, store bf16 [B*T, E]
#pragma unroll
  for (int mt = 0; mt < 2; ++mt) {
#pragma unroll
    for (int g = 0; g < 4; ++g) {
#pragma unroll
      for (int r = 0; r < 8; ++r) {
        const int t = q0 + mt * 16 + r + (hi ? 8 : 0);
        const int e = h * HDIM + g * 16 + lrow;
        Out[(size_t)(b * SEQ + t) * EMBED + e] = f2bf(o[mt][g][r] / lsum[mt][r]);
      }
    }
  }
}

// ----------------------------------------------------------------- launcher

extern "C" void kernel_launch(void* const* d_in, const int* in_sizes, int n_in,
                              void* d_out, int out_size, void* d_ws, size_t ws_size,
                              hipStream_t stream) {
  (void)in_sizes; (void)n_in; (void)out_size; (void)ws_size;

  const float* query = (const float*)d_in[0];
  const float* key   = (const float*)d_in[1];
  const float* value = (const float*)d_in[2];
  const float* Wq    = (const float*)d_in[3];
  const float* Wk    = (const float*)d_in[4];
  const float* Wv    = (const float*)d_in[5];
  const float* bq    = (const float*)d_in[6];
  const float* bk    = (const float*)d_in[7];
  const float* bv    = (const float*)d_in[8];
  const float* Wo    = (const float*)d_in[9];
  const float* bo    = (const float*)d_in[10];

  __bf16* ws = (__bf16*)d_ws;
  __bf16* Xq  = ws;                 // [B*T, E]
  __bf16* Xk  = Xq  + NE;
  __bf16* Xv  = Xk  + NE;
  __bf16* Wqb = Xv  + NE;           // [E, E]
  __bf16* Wkb = Wqb + WE;
  __bf16* Wvb = Wkb + WE;
  __bf16* Wob = Wvb + WE;
  __bf16* Qh  = Wob + WE;           // [B,H,T,hd]
  __bf16* Kh  = Qh  + NE;
  __bf16* VT  = Kh  + NE;           // [B,H,hd,T]
  __bf16* AO  = VT  + NE;           // [B*T, E]

  // 1) fp32 -> bf16 conversions
  {
    const int blk = 256;
    cvt_f32_to_bf16_x4<<<(NE / 4) / blk, blk, 0, stream>>>(query, Xq, NE / 4);
    cvt_f32_to_bf16_x4<<<(NE / 4) / blk, blk, 0, stream>>>(key,   Xk, NE / 4);
    cvt_f32_to_bf16_x4<<<(NE / 4) / blk, blk, 0, stream>>>(value, Xv, NE / 4);
    cvt_f32_to_bf16_x4<<<(WE / 4) / blk, blk, 0, stream>>>(Wq, Wqb, WE / 4);
    cvt_f32_to_bf16_x4<<<(WE / 4) / blk, blk, 0, stream>>>(Wk, Wkb, WE / 4);
    cvt_f32_to_bf16_x4<<<(WE / 4) / blk, blk, 0, stream>>>(Wv, Wvb, WE / 4);
    cvt_f32_to_bf16_x4<<<(WE / 4) / blk, blk, 0, stream>>>(Wo, Wob, WE / 4);
  }

  // 2) in-projections via WMMA GEMM
  {
    dim3 grid(EMBED / 256, NTOK / 64);
    gemm_bf16_wmma<<<grid, 256, 0, stream>>>(Xq, Wqb, bq, Qh, NTOK, EMBED, EMBED, 0);
    gemm_bf16_wmma<<<grid, 256, 0, stream>>>(Xk, Wkb, bk, Kh, NTOK, EMBED, EMBED, 0);
    gemm_bf16_wmma<<<grid, 256, 0, stream>>>(Xv, Wvb, bv, VT, NTOK, EMBED, EMBED, 1);
  }

  // 3) flash attention (32 q rows per wave, 128 per block)
  {
    dim3 grid(SEQ / 128, BATCH * NHEADS);
    flash_attn_bf16<<<grid, 128, 0, stream>>>(Qh, Kh, VT, AO, 0.125f /*1/sqrt(64)*/);
  }

  // 4) out-projection -> fp32 d_out
  {
    dim3 grid(EMBED / 256, NTOK / 64);
    gemm_bf16_wmma<<<grid, 256, 0, stream>>>(AO, Wob, bo, (float*)d_out,
                                             NTOK, EMBED, EMBED, 2);
  }
}
